// ChildSumTreeLSTMCell_80101140070885
// MI455X (gfx1250) — matrix operand order repaired and checked by reading, not compiled, not run on hardware
//
#include <hip/hip_runtime.h>
#include <hip/hip_bf16.h>

// ---------------------------------------------------------------------------
// ChildSumTreeLSTMCell for MI455X (gfx1250, wave32, WMMA + TDM)
//
// Roofline: ~95 GFLOP of GEMM vs ~1 GB HBM traffic (~45us @ 23.3 TB/s).
// bf16 WMMA (v_wmma_f32_16x16x32_bf16, f32 accumulate) balances compute with
// the memory floor; f32 WMMA (K=4) would be ~8x compute-bound.
// GEMM tiles are staged into LDS by the Tensor Data Mover
// (tensor_load_to_lds): the D# pad feature (pad 4 DWORDs every 16 DWORDs)
// reproduces the 80B padded LDS row stride needed for conflict-free
// ds_load_b128 fragment reads. Segment sums use deterministic per-node
// binary-searched ranges (seg_ids sorted) -- no float atomics.
// ---------------------------------------------------------------------------

#define NNODES 16384
#define NEDGES 65536
#define DIN    512
#define HID    512

typedef __attribute__((ext_vector_type(16))) __bf16 v16bf;
typedef __attribute__((ext_vector_type(8)))  __bf16 v8bf;
typedef __attribute__((ext_vector_type(8)))  float  v8f;
typedef __attribute__((ext_vector_type(4)))  unsigned int v4u;
typedef __attribute__((ext_vector_type(8)))  int    v8i_;
typedef __attribute__((ext_vector_type(4)))  int    v4i_;

#if defined(__AMDGCN__) && __has_builtin(__builtin_amdgcn_tensor_load_to_lds) && \
    __has_builtin(__builtin_amdgcn_s_wait_tensorcnt)
#define USE_TDM 1
#else
#define USE_TDM 0
#endif

__device__ __forceinline__ float sigmoidf_(float x) {
    return 1.0f / (1.0f + __expf(-x));
}

// ---------------------------------------------------------------------------
// f32 -> bf16 elementwise convert
// ---------------------------------------------------------------------------
__global__ void f32_to_bf16_kernel(const float* __restrict__ in,
                                   __bf16* __restrict__ out, long n) {
    long i = (long)blockIdx.x * blockDim.x + threadIdx.x;
    if (i < n) out[i] = (__bf16)in[i];
}

// ---------------------------------------------------------------------------
// starts[v] = lower_bound(seg_ids, v); starts[N] = E.  seg_ids sorted.
// ---------------------------------------------------------------------------
__global__ void seg_starts_kernel(const int* __restrict__ seg_ids,
                                  int* __restrict__ starts) {
    int v = blockIdx.x * blockDim.x + threadIdx.x;
    if (v > NNODES) return;
    if (v == NNODES) { starts[NNODES] = NEDGES; return; }
    int lo = 0, hi = NEDGES;
    while (lo < hi) {
        int mid = (lo + hi) >> 1;
        if (seg_ids[mid] < v) lo = mid + 1; else hi = mid;
    }
    starts[v] = lo;
}

// ---------------------------------------------------------------------------
// Build bf16 A_cat[n, 0:512] = x[n,:], A_cat[n, 512:1024] = segsum prev_h.
// ---------------------------------------------------------------------------
__global__ __launch_bounds__(128)
void build_acat_kernel(const float* __restrict__ x,
                       const float* __restrict__ prev_h,
                       const int* __restrict__ starts,
                       __bf16* __restrict__ Acat) {
    const int nid = blockIdx.x;
    const int col = threadIdx.x * 4;
    __bf16* arow = Acat + (size_t)nid * (DIN + HID);

    const float* xr = x + (size_t)nid * DIN + col;
    arow[col + 0] = (__bf16)xr[0];
    arow[col + 1] = (__bf16)xr[1];
    arow[col + 2] = (__bf16)xr[2];
    arow[col + 3] = (__bf16)xr[3];

    const int s0 = starts[nid], s1 = starts[nid + 1];
    float a0 = 0.f, a1 = 0.f, a2 = 0.f, a3 = 0.f;
    for (int e = s0; e < s1; ++e) {
        const float* hr = prev_h + (size_t)e * HID + col;
        a0 += hr[0]; a1 += hr[1]; a2 += hr[2]; a3 += hr[3];
    }
    arow[DIN + col + 0] = (__bf16)a0;
    arow[DIN + col + 1] = (__bf16)a1;
    arow[DIN + col + 2] = (__bf16)a2;
    arow[DIN + col + 3] = (__bf16)a3;
}

// ---------------------------------------------------------------------------
// TDM tile load: 2D tile (tileRows x 32 bf16) from row-major matrix with
// row stride ldElems (bf16 units) -> LDS at ldsAddr with 80B padded rows.
// D# built per CDNA5 ISA ch.8:
//   group0: [1:0]=count=1, [63:32]=lds_addr, [120:64]=global_addr, [127:126]=2
//   group1: data_size=1 (2B), pad_enable=1, pad_interval=3 (16 DW = 64B),
//           pad_amount=3 (4 DW = 16B), tensor_dim0/1 = tile dims,
//           tile_dim0=32, tile_dim1=tileRows, tensor_dim0_stride=ldElems
//   groups 2/3 + trailing group: zero (2D tensor, no gather/iterate)
// Builtin arity: 6-arg form (amdgpu-toolchain clang-23):
//   (uint32x4 g0, int32x8 g1, int32x4, int32x4, int32x8, i32 cpol)
// ---------------------------------------------------------------------------
#if USE_TDM
__device__ __forceinline__ void tdm_load_tile(unsigned ldsAddr,
                                              const __bf16* gptr,
                                              unsigned tileRows,
                                              unsigned ldElems) {
    const unsigned long long ga = (unsigned long long)(uintptr_t)gptr;
    const unsigned tile0 = 32u;

    v4u g0;
    g0[0] = 1u;                                            // count=1, user D#
    g0[1] = ldsAddr;                                       // LDS byte address
    g0[2] = (unsigned)(ga & 0xffffffffu);                  // global_addr[31:0]
    g0[3] = (unsigned)((ga >> 32) & 0x1ffffffu)            // global_addr[56:32]
          | (2u << 30);                                    // type=2 (image)

    const unsigned long long stride = (unsigned long long)ldElems;
    v8i_ g1;
    g1[0] = (int)((1u << 16)                               // data_size = 2B
                | (1u << 20)                               // pad_enable
                | (3u << 22)                               // pad_interval: 16 DW
                | (3u << 25));                             // pad_amount: 4 DW
    g1[1] = (int)((tile0 & 0xffffu) << 16);                // tensor_dim0 lo16
    g1[2] = (int)(((tile0 >> 16) & 0xffffu)                // tensor_dim0 hi16
                | ((tileRows & 0xffffu) << 16));           // tensor_dim1 lo16
    g1[3] = (int)(((tileRows >> 16) & 0xffffu)             // tensor_dim1 hi16
                | ((tile0 & 0xffffu) << 16));              // tile_dim0 = 32
    g1[4] = (int)(tileRows & 0xffffu);                     // tile_dim1; tile_dim2=0
    g1[5] = (int)(stride & 0xffffffffu);                   // dim0_stride[31:0]
    g1[6] = (int)((stride >> 32) & 0xffffu);               // dim0_stride[47:32]
    g1[7] = 0;                                             // dim1_stride unused
    v4i_ gz4 = {0, 0, 0, 0};
    v8i_ gz8 = {0, 0, 0, 0, 0, 0, 0, 0};
    __builtin_amdgcn_tensor_load_to_lds(g0, g1, gz4, gz4, gz8, 0);
}
#endif

// ---------------------------------------------------------------------------
// Tiled bf16 WMMA GEMM:  C[m,n] = sum_k A[m,k]*B[n,k] + bias[n]
//   block tile 128(M) x 64(N), K stepped by 32 through LDS (row pad 8 bf16
//   -> 80B stride, conflict-free b128 reads). 8 waves: 4(M) x 2(N), each
//   wave owns 32x32 = 2x2 v_wmma_f32_16x16x32_bf16 fragments.
// FUSE epilogue (f_hiddens path):
//   C[e,n] = sigmoid(acc + bias[n] + f_in[seg_ids[e], n]) * prev_c[e, n]
// Fragment layouts per CDNA5 ISA 7.12.2:
//   A (16x32, MxK): lane<16 holds M=lane, K {0..7,16..23}; lane>=16 K {8..15,24..31}
//   B (32x16, KxN): lane<16 holds N=lane, K {0..7,16..23}; lane>=16 K {8..15,24..31}
// ---------------------------------------------------------------------------
#define LDS_STRIDE 40   // 32 bf16 payload + 8 bf16 pad (= 80B = TDM pad)

template <bool FUSE>
__global__ __launch_bounds__(256)
void gemm_bf16_wmma(const __bf16* __restrict__ A, int lda,
                    const __bf16* __restrict__ B, int ldb,
                    const float* __restrict__ bias,
                    float* __restrict__ C, int ldc, int K,
                    const int*   __restrict__ seg_ids,
                    const float* __restrict__ f_in,
                    const float* __restrict__ prev_c) {
    __shared__ __bf16 As[128 * LDS_STRIDE];
    __shared__ __bf16 Bs[64 * LDS_STRIDE];

    const int mBase = blockIdx.x * 128;
    const int nBase = blockIdx.y * 64;
    const int t     = threadIdx.x;
    const int lane  = t & 31;
    const int wid   = t >> 5;
    const int waveM = wid >> 1;      // 0..3
    const int waveN = wid & 1;       // 0..1
    const int laneM = lane & 15;
    const int half  = lane >> 4;     // 0|1

    v8f acc[2][2];
    #pragma unroll
    for (int mi = 0; mi < 2; ++mi)
        #pragma unroll
        for (int ni = 0; ni < 2; ++ni)
            #pragma unroll
            for (int q = 0; q < 8; ++q) acc[mi][ni][q] = 0.0f;

    for (int kt = 0; kt < K; kt += 32) {
#if USE_TDM
        // ---- TDM staging: wave 0 DMAs both tiles, waits TENSORcnt ----
        if (wid == 0) {
            tdm_load_tile((unsigned)(uintptr_t)&As[0],
                          A + (size_t)mBase * lda + kt, 128u, (unsigned)lda);
            tdm_load_tile((unsigned)(uintptr_t)&Bs[0],
                          B + (size_t)nBase * ldb + kt, 64u, (unsigned)ldb);
            __builtin_amdgcn_s_wait_tensorcnt(0);
        }
        __syncthreads();
#else
        // ---- manual staging fallback ----
        #pragma unroll
        for (int it = 0; it < 2; ++it) {
            int i = t + it * 256;
            int r = i >> 2, c = (i & 3) << 3;
            *(v8bf*)(&As[r * LDS_STRIDE + c]) =
                *(const v8bf*)(A + (size_t)(mBase + r) * lda + kt + c);
        }
        {
            int r = t >> 2, c = (t & 3) << 3;
            *(v8bf*)(&Bs[r * LDS_STRIDE + c]) =
                *(const v8bf*)(B + (size_t)(nBase + r) * ldb + kt + c);
        }
        __syncthreads();
#endif

        v16bf af[2], bfr[2];
        const int ac0 = half * 8;    // A: K groups of 8, swapped across lane halves
        #pragma unroll
        for (int mi = 0; mi < 2; ++mi) {
            const __bf16* p = &As[(waveM * 32 + mi * 16 + laneM) * LDS_STRIDE];
            v8bf lo = *(const v8bf*)(p + ac0);
            v8bf hi = *(const v8bf*)(p + ac0 + 16);
            #pragma unroll
            for (int q = 0; q < 8; ++q) { af[mi][q] = lo[q]; af[mi][q + 8] = hi[q]; }
        }
        const int bk0 = half * 8;    // B: K groups of 8, swapped across lane halves (same as A)
        #pragma unroll
        for (int ni = 0; ni < 2; ++ni) {
            const __bf16* p = &Bs[(waveN * 32 + ni * 16 + laneM) * LDS_STRIDE];
            v8bf lo = *(const v8bf*)(p + bk0);
            v8bf hi = *(const v8bf*)(p + bk0 + 16);
            #pragma unroll
            for (int q = 0; q < 8; ++q) { bfr[ni][q] = lo[q]; bfr[ni][q + 8] = hi[q]; }
        }

        #pragma unroll
        for (int mi = 0; mi < 2; ++mi)
            #pragma unroll
            for (int ni = 0; ni < 2; ++ni)
                acc[mi][ni] = __builtin_amdgcn_wmma_f32_16x16x32_bf16(
                    false, af[mi], false, bfr[ni],
                    (short)0, acc[mi][ni], false, false);

        __syncthreads();
    }

    // ---- epilogue ----
    #pragma unroll
    for (int mi = 0; mi < 2; ++mi) {
        #pragma unroll
        for (int ni = 0; ni < 2; ++ni) {
            const int colT = nBase + waveN * 32 + ni * 16 + laneM;
            const float bv = bias[colT];
            #pragma unroll
            for (int q = 0; q < 8; ++q) {
                const int rowT = mBase + waveM * 32 + mi * 16 + half * 8 + q;
                float v = acc[mi][ni][q] + bv;
                if (FUSE) {
                    const int seg = seg_ids[rowT];
                    v = sigmoidf_(v + f_in[(size_t)seg * ldc + colT]);
                    v *= prev_c[(size_t)rowT * ldc + colT];
                }
                C[(size_t)rowT * ldc + colT] = v;
            }
        }
    }
}

// ---------------------------------------------------------------------------
// Final: fc = segsum(s), c = sig(z_i)*tanh(z_u)+fc, h = sig(z_o)*tanh(c)
// ---------------------------------------------------------------------------
__global__ __launch_bounds__(128)
void finalize_kernel(const float* __restrict__ big,
                     const float* __restrict__ s,
                     const int* __restrict__ starts,
                     float* __restrict__ out) {
    const int nid = blockIdx.x;
    const int col = threadIdx.x * 4;
    const int s0 = starts[nid], s1 = starts[nid + 1];

    float f0 = 0.f, f1 = 0.f, f2 = 0.f, f3 = 0.f;
    for (int e = s0; e < s1; ++e) {
        const float* sr = s + (size_t)e * HID + col;
        f0 += sr[0]; f1 += sr[1]; f2 += sr[2]; f3 += sr[3];
    }
    const float* br = big + (size_t)nid * (3 * HID);
    #pragma unroll
    for (int j = 0; j < 4; ++j) {
        const float zi = br[col + j];
        const float zo = br[HID + col + j];
        const float zu = br[2 * HID + col + j];
        float fc = (j == 0) ? f0 : (j == 1) ? f1 : (j == 2) ? f2 : f3;
        const float c = sigmoidf_(zi) * tanhf(zu) + fc;
        const float h = sigmoidf_(zo) * tanhf(c);
        out[(size_t)nid * HID + col + j] = c;
        out[(size_t)NNODES * HID + (size_t)nid * HID + col + j] = h;
    }
}

// ---------------------------------------------------------------------------
// Launcher
// ---------------------------------------------------------------------------
extern "C" void kernel_launch(void* const* d_in, const int* in_sizes, int n_in,
                              void* d_out, int out_size, void* d_ws, size_t ws_size,
                              hipStream_t stream) {
    (void)in_sizes; (void)n_in; (void)out_size; (void)ws_size;

    const float* x       = (const float*)d_in[0];
    const float* prev_c  = (const float*)d_in[1];
    const float* prev_h  = (const float*)d_in[2];
    const int*   seg_ids = (const int*)d_in[3];
    const float* Wc      = (const float*)d_in[4];
    const float* bc      = (const float*)d_in[5];
    const float* Wwf     = (const float*)d_in[6];
    const float* bwf     = (const float*)d_in[7];
    const float* Wuf     = (const float*)d_in[8];
    const float* buf     = (const float*)d_in[9];
    float* out = (float*)d_out;

    // workspace carve-out (256B aligned)
    char* ws = (char*)d_ws;
    size_t off = 0;
    auto carve = [&](size_t bytes) -> char* {
        char* p = ws + off;
        off += (bytes + 255) & ~(size_t)255;
        return p;
    };
    int*    starts  = (int*)   carve((NNODES + 1) * sizeof(int));
    __bf16* Acat    = (__bf16*)carve((size_t)NNODES * (DIN + HID) * 2);
    __bf16* prevhb  = (__bf16*)carve((size_t)NEDGES * HID * 2);
    __bf16* Wcb     = (__bf16*)carve((size_t)3 * HID * (DIN + HID) * 2);
    __bf16* Wwfb    = (__bf16*)carve((size_t)HID * DIN * 2);
    __bf16* Wufb    = (__bf16*)carve((size_t)HID * HID * 2);
    float*  big     = (float*) carve((size_t)NNODES * 3 * HID * 4);
    float*  f_in    = (float*) carve((size_t)NNODES * HID * 4);
    float*  s_buf   = (float*) carve((size_t)NEDGES * HID * 4);

    // 1) bf16 conversions
    {
        long n;
        n = (long)3 * HID * (DIN + HID);
        f32_to_bf16_kernel<<<(n + 255) / 256, 256, 0, stream>>>(Wc, Wcb, n);
        n = (long)HID * DIN;
        f32_to_bf16_kernel<<<(n + 255) / 256, 256, 0, stream>>>(Wwf, Wwfb, n);
        n = (long)HID * HID;
        f32_to_bf16_kernel<<<(n + 255) / 256, 256, 0, stream>>>(Wuf, Wufb, n);
        n = (long)NEDGES * HID;
        f32_to_bf16_kernel<<<(n + 255) / 256, 256, 0, stream>>>(prev_h, prevhb, n);
    }

    // 2) segment offsets
    seg_starts_kernel<<<(NNODES + 1 + 255) / 256, 256, 0, stream>>>(seg_ids, starts);

    // 3) h_tilde segsum + bf16 concat [x | h_tilde]
    build_acat_kernel<<<NNODES, 128, 0, stream>>>(x, prev_h, starts, Acat);

    // 4) big = Acat @ Wc^T + bc        [16384 x 1536], K=1024
    gemm_bf16_wmma<false><<<dim3(NNODES / 128, (3 * HID) / 64), 256, 0, stream>>>(
        Acat, DIN + HID, Wcb, DIN + HID, bc, big, 3 * HID, DIN + HID,
        nullptr, nullptr, nullptr);

    // 5) f_in = x @ Wwf^T + bwf        [16384 x 512], K=512 (x lives in Acat cols 0..511)
    gemm_bf16_wmma<false><<<dim3(NNODES / 128, HID / 64), 256, 0, stream>>>(
        Acat, DIN + HID, Wwfb, DIN, bwf, f_in, HID, DIN,
        nullptr, nullptr, nullptr);

    // 6) s = sigmoid(prev_h @ Wuf^T + buf + f_in[seg]) * prev_c   [65536 x 512], K=512
    gemm_bf16_wmma<true><<<dim3(NEDGES / 128, HID / 64), 256, 0, stream>>>(
        prevhb, HID, Wufb, HID, buf, s_buf, HID, HID,
        seg_ids, f_in, prev_c);

    // 7) fc segsum + gates -> (c, h)
    finalize_kernel<<<NNODES, 128, 0, stream>>>(big, s_buf, starts, out);
}
